// GraphAttention_40295383171688
// MI455X (gfx1250) — matrix operand order repaired and checked by reading, not compiled, not run on hardware
//
#include <hip/hip_runtime.h>
#include <hip/hip_bf16.h>

// ---------------------------------------------------------------------------
// GAT-style kernel for MI455X (gfx1250, wave32).
//   N=8192 nodes, F=512 in-features, H=2 heads, F_=128 out-features/head.
// Design: never materialize the dense [H,N,N] attention matrix. A is 0/1 with
// self loops, so mask_pos == A; attention rows have ~33 nonzeros. We stream
// A once (268 MB = the HBM roofline term @ 23.3 TB/s ~ 11.5 us), evaluate
// exp(exp(-(s-t)^2)) only at nonzeros, and gather feat rows (8 MB, L2-
// resident) for aggregation. The X@W projection uses v_wmma_f32_16x16x32_f16
// with W pre-packed into a contiguous-k B-fragment layout (same k order as A) (f16) and a one-deep
// software pipeline so fragment loads overlap the WMMAs.
// ---------------------------------------------------------------------------

#define GN 8192
#define GF 512
#define GH 2
#define GD 128

typedef __attribute__((ext_vector_type(16))) _Float16 v16h;
typedef __attribute__((ext_vector_type(8)))  float    v8f;

// ---------------------------------------------------------------------------
// Kernel 0: pack W[h,f,d] (f32) into f16 WMMA B-fragments.
// Fragment id = (head*16 + Kstep)*8 + tile; per fragment: 32 lanes x 16 halves.
// B packing (16-bit 32x16, contiguous k as in A): lane l -> col = tile*16 + (l&15),
//                          half i -> k  = Kstep*32 + (l>>4)*16 + i.
// ---------------------------------------------------------------------------
__global__ __launch_bounds__(32) void wpack_kernel(const float* __restrict__ W,
                                                   _Float16* __restrict__ Wp) {
    const int id = blockIdx.x;            // 256 fragments
    const int l  = threadIdx.x;
    const int h  = id >> 7;
    const int ks = (id >> 3) & 15;
    const int tt = id & 7;
    const int col = tt * 16 + (l & 15);
    const int kb  = ks * 32 + (l >> 4) * 16;
    const float* Wh = W + (size_t)h * GF * GD;
    _Float16* dst = Wp + ((size_t)id * 32 + l) * 16;
#pragma unroll
    for (int i = 0; i < 16; ++i)
        dst[i] = (_Float16)Wh[(size_t)(kb + i) * GD + col];
}

// ---------------------------------------------------------------------------
// Kernel 1: feat[h,n,d] = sum_f X[n,f] * W[h,f,d]
// One wave computes a 16-row x 128-col strip for one head.
// waves = H * (N/16) = 1024; 8 waves/block -> 128 blocks.
// Software-pipelined: B fragment tt+1 and the next K-step's A fragment are
// loaded while WMMA tt executes.
// ---------------------------------------------------------------------------
__device__ __forceinline__ v16h cvt_a_frag(const float4* Xr4, int ks, int hi) {
    const int q0 = ks * 8 + hi * 4;         // float4 index of k0 = ks*32 + 16*hi
    const float4 p0 = Xr4[q0];
    const float4 p1 = Xr4[q0 + 1];
    const float4 p2 = Xr4[q0 + 2];
    const float4 p3 = Xr4[q0 + 3];
    v16h a;
    a[0]  = (_Float16)p0.x; a[1]  = (_Float16)p0.y;
    a[2]  = (_Float16)p0.z; a[3]  = (_Float16)p0.w;
    a[4]  = (_Float16)p1.x; a[5]  = (_Float16)p1.y;
    a[6]  = (_Float16)p1.z; a[7]  = (_Float16)p1.w;
    a[8]  = (_Float16)p2.x; a[9]  = (_Float16)p2.y;
    a[10] = (_Float16)p2.z; a[11] = (_Float16)p2.w;
    a[12] = (_Float16)p3.x; a[13] = (_Float16)p3.y;
    a[14] = (_Float16)p3.z; a[15] = (_Float16)p3.w;
    return a;
}

__global__ __launch_bounds__(256) void feat_kernel(const float* __restrict__ X,
                                                   const _Float16* __restrict__ Wp,
                                                   float* __restrict__ feat) {
    const int l    = threadIdx.x & 31;
    const int wid  = blockIdx.x * 8 + (threadIdx.x >> 5);
    const int head = wid >> 9;              // 0..1
    const int m0   = (wid & 511) << 4;      // row tile * 16

    const int rl  = l & 15;                 // sub-lane position
    const int hi  = l >> 4;                 // half-wave select
    const int row = m0 + rl;
    const float4* Xr4 = reinterpret_cast<const float4*>(X + (size_t)row * GF);

    // Fragment (ks, tt) lives at wb + (ks*8 + tt) * 512 halves for this lane.
    const _Float16* wb = Wp + (((size_t)head * 16 * 8) * 32 + l) * 16;
    auto bfrag = [&](int ks, int tt) -> v16h {
        return *reinterpret_cast<const v16h*>(wb + (size_t)(ks * 8 + tt) * 32 * 16);
    };

    v8f acc[8] = {};                        // 8 N-tiles of 16 cols

    v16h a    = cvt_a_frag(Xr4, 0, hi);     // pipeline prologue
    v16h bcur = bfrag(0, 0);

    for (int ks = 0; ks < 16; ++ks) {
        const v16h anext = (ks < 15) ? cvt_a_frag(Xr4, ks + 1, hi) : a;
#pragma unroll
        for (int tt = 0; tt < 8; ++tt) {
            v16h bnext;
            if (tt < 7)          bnext = bfrag(ks, tt + 1);
            else if (ks < 15)    bnext = bfrag(ks + 1, 0);
            else                 bnext = bcur;
            acc[tt] = __builtin_amdgcn_wmma_f32_16x16x32_f16(
                false, a, false, bcur, (short)0, acc[tt], false, false);
            bcur = bnext;
        }
        a = anext;
    }

    // C/D layout: VGPR r -> M = r + 8*hi, N = rl (per 16-col tile)
#pragma unroll
    for (int tt = 0; tt < 8; ++tt) {
        const int col = tt * 16 + rl;
#pragma unroll
        for (int r = 0; r < 8; ++r) {
            const int orow = m0 + r + 8 * hi;
            feat[((size_t)head * GN + orow) * GD + col] = acc[tt][r];
        }
    }
}

// ---------------------------------------------------------------------------
// Kernel 2: s[h,n] = feat[h,n,:].a_self[h],  t[h,n] = feat[h,n,:].a_neigh[h]
// One wave per node.
// ---------------------------------------------------------------------------
__global__ __launch_bounds__(32) void st_kernel(const float* __restrict__ feat,
                                                const float* __restrict__ a_self,
                                                const float* __restrict__ a_neigh,
                                                float* __restrict__ sarr,
                                                float* __restrict__ tarr) {
    const int n = blockIdx.x;
    const int l = threadIdx.x;
#pragma unroll
    for (int h = 0; h < GH; ++h) {
        const float* fr = feat + ((size_t)h * GN + n) * GD;
        float ps = 0.f, pt = 0.f;
#pragma unroll
        for (int r = 0; r < 4; ++r) {
            const int d = l + 32 * r;
            const float v = fr[d];
            ps += v * a_self[h * GD + d];
            pt += v * a_neigh[h * GD + d];
        }
#pragma unroll
        for (int o = 16; o > 0; o >>= 1) {
            ps += __shfl_down(ps, o, 32);
            pt += __shfl_down(pt, o, 32);
        }
        if (l == 0) { sarr[h * GN + n] = ps; tarr[h * GN + n] = pt; }
    }
}

// ---------------------------------------------------------------------------
// Kernel 3: sparse-aware row aggregation.
// One wave per node i: scan A[i,:] (float4-coalesced, prefetch next chunk),
// compact nonzero columns into an LDS queue, then for each neighbor j:
//   w_h = exp(exp(-(s_h[i]-t_h[j])^2));  acc_h += w_h * feat[h,j,:]
// Finish with row-normalize, +bias, ReLU into out[i, h*128+d].
// ---------------------------------------------------------------------------
__global__ __launch_bounds__(32) void agg_kernel(const float* __restrict__ A,
                                                 const float* __restrict__ feat,
                                                 const float* __restrict__ sarr,
                                                 const float* __restrict__ tarr,
                                                 const float* __restrict__ bias,
                                                 float* __restrict__ out) {
    __shared__ int q[512];
    __shared__ int qn;

    const int i = blockIdx.x;
    const int l = threadIdx.x;

    const float s0 = sarr[i];
    const float s1 = sarr[GN + i];

    float acc0[4] = {0.f, 0.f, 0.f, 0.f};
    float acc1[4] = {0.f, 0.f, 0.f, 0.f};
    float sum0 = 0.f, sum1 = 0.f;

    const float4* Arow = reinterpret_cast<const float4*>(A + (size_t)i * GN);
    if (l == 0) qn = 0;
    __syncthreads();

    // 16 phases x 512 columns (128 float4 per phase, 4 float4 per lane)
    for (int phase = 0; phase < 16; ++phase) {
        const int base = phase * 128;
        if (phase + 1 < 16)  // stream the next A chunk toward the caches
            __builtin_prefetch((const void*)(Arow + base + 128 + l), 0, 1);
#pragma unroll
        for (int u = 0; u < 4; ++u) {
            const int fidx = base + l + 32 * u;     // coalesced float4 index
            const float4 v = Arow[fidx];
            const int c = fidx * 4;
            if (v.x != 0.f) { q[atomicAdd(&qn, 1)] = c; }
            if (v.y != 0.f) { q[atomicAdd(&qn, 1)] = c + 1; }
            if (v.z != 0.f) { q[atomicAdd(&qn, 1)] = c + 2; }
            if (v.w != 0.f) { q[atomicAdd(&qn, 1)] = c + 3; }
        }
        __syncthreads();
        const int m = qn;
        for (int e = 0; e < m; ++e) {
            const int j = q[e];
            const float d0 = s0 - tarr[j];
            const float d1 = s1 - tarr[GN + j];
            const float w0 = __expf(__expf(-d0 * d0));   // exp(dense) as in ref
            const float w1 = __expf(__expf(-d1 * d1));
            sum0 += w0; sum1 += w1;
            const float* f0 = feat + (size_t)j * GD;             // head 0
            const float* f1 = feat + ((size_t)GN + j) * GD;      // head 1
#pragma unroll
            for (int r = 0; r < 4; ++r) {
                const int d = l + 32 * r;
                acc0[r] = fmaf(w0, f0[d], acc0[r]);
                acc1[r] = fmaf(w1, f1[d], acc1[r]);
            }
        }
        __syncthreads();
        if (l == 0) qn = 0;
        __syncthreads();
    }

    const float inv0 = 1.f / sum0;      // self-loop guarantees sum > 0
    const float inv1 = 1.f / sum1;
    float* orow = out + (size_t)i * (GH * GD);
#pragma unroll
    for (int r = 0; r < 4; ++r) {
        const int d = l + 32 * r;
        orow[d]      = fmaxf(acc0[r] * inv0 + bias[d],      0.f);
        orow[GD + d] = fmaxf(acc1[r] * inv1 + bias[GD + d], 0.f);
    }
}

// ---------------------------------------------------------------------------
extern "C" void kernel_launch(void* const* d_in, const int* in_sizes, int n_in,
                              void* d_out, int out_size, void* d_ws, size_t ws_size,
                              hipStream_t stream) {
    const float* X       = (const float*)d_in[0];   // [N,F]
    const float* A       = (const float*)d_in[1];   // [N,N]
    const float* W       = (const float*)d_in[2];   // [H,F,F_]
    const float* b       = (const float*)d_in[3];   // [H,F_]
    const float* a_self  = (const float*)d_in[4];   // [H,F_]
    const float* a_neigh = (const float*)d_in[5];   // [H,F_]
    float* out = (float*)d_out;                     // [N, H*F_]

    // Workspace layout: feat [H,N,F_] (8 MB) + s,t [H,N] + packed W (f16).
    float* feat = (float*)d_ws;
    float* s    = feat + (size_t)GH * GN * GD;
    float* t    = s + (size_t)GH * GN;
    _Float16* Wp = (_Float16*)(t + (size_t)GH * GN);   // 256 KB, 32B-aligned

    wpack_kernel<<<256, 32, 0, stream>>>(W, Wp);
    feat_kernel<<<128, 256, 0, stream>>>(X, Wp, feat);
    st_kernel<<<GN, 32, 0, stream>>>(feat, a_self, a_neigh, s, t);
    agg_kernel<<<GN, 32, 0, stream>>>(A, feat, s, t, b, out);
}
